// DilatedAttention_54752243089749
// MI455X (gfx1250) — compile-verified, matrix operand order checked by reading, not run on hardware
//
#include <hip/hip_runtime.h>

typedef __attribute__((ext_vector_type(16))) __bf16 v16bf;
typedef __attribute__((ext_vector_type(8)))  __bf16 v8bf;
typedef __attribute__((ext_vector_type(8)))  float  v8f;

#define XPITCH 1032   // 1024 + 8 bf16 pad -> 2064B row stride, 16B aligned, bank-spread
#define VPITCH 40     // v^T rows: 32 + 8 pad
#define APITCH 40     // attn rows: 32 + 8 pad
#define SPITCH 33     // sim fp32 rows

// Load a 16x32 (MxK) A-fragment or 32x16 (KxN) B-fragment whose "index" rows
// are contiguous in memory (A: row m contiguous in K; B: column n == W row n
// contiguous in K). CDNA5 16-bit WMMA striping:
//   lane<16 : elems 0..7 = K k0+0..7,   elems 8..15 = K k0+16..23
//   lane>=16: elems 0..7 = K k0+8..15,  elems 8..15 = K k0+24..31
__device__ __forceinline__ v16bf load_frag(const __bf16* __restrict__ base,
                                           int idx0, int pitch, int k0, int lane) {
  const int idx  = lane & 15;
  const int half = lane >> 4;
  const __bf16* p = base + (size_t)(idx0 + idx) * pitch + k0 + half * 8;
  union { v16bf v; v8bf h[2]; } u;
  u.h[0] = *(const v8bf*)(p);
  u.h[1] = *(const v8bf*)(p + 16);
  return u.v;
}

__device__ __forceinline__ v8f wmma_bf16(v16bf a, v16bf b, v8f c) {
  return __builtin_amdgcn_wmma_f32_16x16x32_bf16(false, a, false, b, (short)0, c,
                                                 false, false);
}

// Store one 16x16 f32 accumulator tile (+bias) to LDS as bf16.
template <bool TRANSPOSE>
__device__ __forceinline__ void store_tile(__bf16* __restrict__ dst, v8f acc,
                                           int m0, int n0,
                                           const float* __restrict__ bias, int lane) {
  const int col   = n0 + (lane & 15);
  const int rbase = m0 + ((lane >> 4) << 3);
  const float bv  = bias[col];
  if (TRANSPOSE) {
    for (int r = 0; r < 8; ++r)
      dst[(size_t)col * VPITCH + rbase + r] = (__bf16)(acc[r] + bv);
  } else {
    for (int r = 0; r < 8; ++r)
      dst[(size_t)(rbase + r) * XPITCH + col] = (__bf16)(acc[r] + bv);
  }
}

// One projection: dst = x_seg(32x1024) @ W^T + bias. Per wave: 8 N-tiles as
// 4 pairs; 2 M-tiles x 2 N-tiles register tile -> 4 WMMAs per 4 frag loads.
template <bool TRANSPOSE>
__device__ __forceinline__ void projection(const __bf16* __restrict__ xs,
                                           const __bf16* __restrict__ W,
                                           const float* __restrict__ bias,
                                           __bf16* __restrict__ dst,
                                           int wave, int lane) {
  for (int jj = 0; jj < 4; ++jj) {
    const int n0a = (wave + 16 * jj) * 16;
    const int n0b = n0a + 128;                    // +8 tiles
    v8f a00 = {}, a10 = {}, a01 = {}, a11 = {};
    for (int kc = 0; kc < 1024; kc += 32) {
      v16bf fa0 = load_frag(xs, 0,  XPITCH, kc, lane);
      v16bf fa1 = load_frag(xs, 16, XPITCH, kc, lane);
      v16bf fb0 = load_frag(W, n0a, 1024, kc, lane);   // W row n == B col n
      v16bf fb1 = load_frag(W, n0b, 1024, kc, lane);
      a00 = wmma_bf16(fa0, fb0, a00);
      a10 = wmma_bf16(fa1, fb0, a10);
      a01 = wmma_bf16(fa0, fb1, a01);
      a11 = wmma_bf16(fa1, fb1, a11);
    }
    store_tile<TRANSPOSE>(dst, a00, 0,  n0a, bias, lane);
    store_tile<TRANSPOSE>(dst, a10, 16, n0a, bias, lane);
    store_tile<TRANSPOSE>(dst, a01, 0,  n0b, bias, lane);
    store_tile<TRANSPOSE>(dst, a11, 16, n0b, bias, lane);
  }
}

__global__ void convert_bf16_kernel(const float* __restrict__ src,
                                    __bf16* __restrict__ dst, int n) {
  int i = (blockIdx.x * blockDim.x + threadIdx.x) * 4;
  if (i + 3 < n) {
    float4 f = *(const float4*)(src + i);
    dst[i + 0] = (__bf16)f.x;
    dst[i + 1] = (__bf16)f.y;
    dst[i + 2] = (__bf16)f.z;
    dst[i + 3] = (__bf16)f.w;
  }
}

__global__ __launch_bounds__(256)
void dilated_attn_kernel(const float* __restrict__ x,
                         const __bf16* __restrict__ Wq,
                         const __bf16* __restrict__ Wk,
                         const __bf16* __restrict__ Wv,
                         const float* __restrict__ bq,
                         const float* __restrict__ bk,
                         const float* __restrict__ bv,
                         float* __restrict__ out) {
  const int seg  = blockIdx.x;        // 0..511 = b*128 + s
  const int b    = seg >> 7;
  const int s    = seg & 127;
  const int tid  = threadIdx.x;
  const int wave = tid >> 5;
  const int lane = tid & 31;

  extern __shared__ char smem[];
  __bf16* xs  = (__bf16*)smem;                    // 32 x XPITCH
  __bf16* qs  = xs + 32 * XPITCH;                 // 32 x XPITCH
  __bf16* ks  = qs + 32 * XPITCH;                 // 32 x XPITCH
  __bf16* vst = ks + 32 * XPITCH;                 // 1024 x VPITCH (v transposed)
  float*  sim  = (float*)xs;                      // overlay (xs dead after proj)
  __bf16* attn = (__bf16*)(sim + 32 * SPITCH);    // overlay

  // ---- 1) dilated load: rows 0,2,..,62 of segment -> bf16 LDS ----
  const float* xbase = x + ((size_t)(b * 8192 + s * 64)) * 1024;
  for (int e = tid * 4; e < 32 * 1024; e += 256 * 4) {
    int row = e >> 10, col = e & 1023;
    float4 f = *(const float4*)(xbase + (size_t)(row * 2) * 1024 + col);
    __bf16* d = xs + row * XPITCH + col;
    d[0] = (__bf16)f.x; d[1] = (__bf16)f.y; d[2] = (__bf16)f.z; d[3] = (__bf16)f.w;
  }
  __syncthreads();

  // ---- 2) Q/K/V projections (distinct pointer chains -> global_load) ----
  projection<false>(xs, Wq, bq, qs,  wave, lane);
  projection<false>(xs, Wk, bk, ks,  wave, lane);
  projection<true >(xs, Wv, bv, vst, wave, lane);
  __syncthreads();

  // ---- 3) sim = (q @ k^T) * D^-1/2 : 2x2 tiles, waves 0..3 ----
  if (wave < 4) {
    int m0 = (wave & 1) * 16, n0 = (wave >> 1) * 16;
    v8f acc = {};
    for (int kc = 0; kc < 1024; kc += 32) {
      v16bf a  = load_frag(qs, m0, XPITCH, kc, lane);
      v16bf bf = load_frag(ks, n0, XPITCH, kc, lane);  // k row j == B col j
      acc = wmma_bf16(a, bf, acc);
    }
    int col = n0 + (lane & 15), rbase = m0 + ((lane >> 4) << 3);
    for (int r = 0; r < 8; ++r)
      sim[(rbase + r) * SPITCH + col] = acc[r] * 0.03125f;  // 1/sqrt(1024)
  }
  __syncthreads();

  // ---- 4) row softmax (32x32), wave 0, one row per lane ----
  if (tid < 32) {
    const float* row = sim + tid * SPITCH;
    float mx = -3.4e38f;
    for (int j = 0; j < 32; ++j) mx = fmaxf(mx, row[j]);
    float sum = 0.f;
    for (int j = 0; j < 32; ++j) sum += __expf(row[j] - mx);
    float inv = 1.f / sum;
    for (int j = 0; j < 32; ++j)
      attn[tid * APITCH + j] = (__bf16)(__expf(row[j] - mx) * inv);
  }
  __syncthreads();

  // ---- 5) out = attn @ v : single-K-chunk WMMA per 16x16 tile ----
  float* obase = out + ((size_t)(b * 4096 + s * 32)) * 1024;
  for (int t = wave; t < 128; t += 8) {
    int m0 = (t & 1) * 16, n0 = (t >> 1) * 16;
    v16bf a  = load_frag(attn, m0, APITCH, 0, lane);
    v16bf bf = load_frag(vst,  n0, VPITCH, 0, lane);  // v^T row n == B col n
    v8f acc = {};
    acc = wmma_bf16(a, bf, acc);
    int col = n0 + (lane & 15), rbase = m0 + ((lane >> 4) << 3);
    for (int r = 0; r < 8; ++r)
      obase[(size_t)(rbase + r) * 1024 + col] = acc[r];
  }
}

extern "C" void kernel_launch(void* const* d_in, const int* in_sizes, int n_in,
                              void* d_out, int out_size, void* d_ws, size_t ws_size,
                              hipStream_t stream) {
  const float* x  = (const float*)d_in[0];
  const float* Wq = (const float*)d_in[1];
  const float* bq = (const float*)d_in[2];
  const float* Wk = (const float*)d_in[3];
  const float* bk = (const float*)d_in[4];
  const float* Wv = (const float*)d_in[5];
  const float* bv = (const float*)d_in[6];
  float* out = (float*)d_out;

  const int WN = 1024 * 1024;
  __bf16* wqb = (__bf16*)d_ws;          // 2 MB each, 6 MB total in d_ws
  __bf16* wkb = wqb + WN;
  __bf16* wvb = wkb + WN;

  convert_bf16_kernel<<<WN / (256 * 4), 256, 0, stream>>>(Wq, wqb, WN);
  convert_bf16_kernel<<<WN / (256 * 4), 256, 0, stream>>>(Wk, wkb, WN);
  convert_bf16_kernel<<<WN / (256 * 4), 256, 0, stream>>>(Wv, wvb, WN);

  // dynamic LDS: 3*(32*1032) + 1024*40 bf16 = 280,064 B  (<= 320 KB WGP LDS)
  size_t smem = (size_t)(3 * 32 * XPITCH + 1024 * VPITCH) * sizeof(__bf16);
  dilated_attn_kernel<<<512, 256, smem, stream>>>(x, wqb, wkb, wvb, bq, bk, bv, out);
}
